// Attention_90417651516180
// MI455X (gfx1250) — matrix operand order repaired and checked by reading, not compiled, not run on hardware
//
#include <hip/hip_runtime.h>
#include <hip/hip_bf16.h>

// ---------------------------------------------------------------------------
// MI455X (gfx1250) attention, bf16 WMMA with f32 accumulation.
//   q,k,v = x @ W^T ; scores = q k^T / 32 ; softmax ; ctx = attn v ; out = ctx wo^T
// GEMM: block tile 128x128xBK32, 8 waves (2M x 4N), wave tile 64x32 = 8 WMMA/step.
// Global->LDS staging uses CDNA5 async copies (global_load_async_to_lds_b128,
// ASYNCcnt) with double-buffered LDS.
// ---------------------------------------------------------------------------

typedef __attribute__((ext_vector_type(16))) __bf16    bf16x16;
typedef __attribute__((ext_vector_type(8)))  float     f32x8;
typedef __attribute__((ext_vector_type(4)))  unsigned  u32x4;

union ABFrag { bf16x16 bf; u32x4 u[2]; };

__device__ __forceinline__ unsigned short f2bf(float f) {
    unsigned u = __float_as_uint(f);
    u += 0x7FFFu + ((u >> 16) & 1u);          // round-to-nearest-even
    return (unsigned short)(u >> 16);
}

// CDNA5 async global -> LDS copy (16B per lane), tracked by ASYNCcnt.
// VDST = per-lane LDS byte address, VADDR = per-lane 64-bit global address.
__device__ __forceinline__ void g2l_async_b128(unsigned lds_off, const void* g) {
    asm volatile("global_load_async_to_lds_b128 %0, %1, off"
                 :: "v"(lds_off), "v"(g) : "memory");
}
__device__ __forceinline__ void wait_async0() {
    asm volatile("s_wait_asynccnt 0x0" ::: "memory");
}

// -------------------------- fp32 -> bf16 convert ---------------------------
__global__ __launch_bounds__(256) void cvt_f32_bf16(const float* __restrict__ in,
                                                    unsigned short* __restrict__ out,
                                                    int n) {
    int i = blockIdx.x * 256 + threadIdx.x;
    if (i < n) out[i] = f2bf(in[i]);
}

// ------------------------------ WMMA GEMM ----------------------------------
// C[M,N] = scale * A[M,K] * op(B)
//   TRANSB=1 : B stored [N,K] -> C = A B^T   (async b128 staging)
//   TRANSB=0 : B stored [K,N] -> C = A B     (sync scatter-transpose staging)
#define BM 128
#define BN 128
#define BK 32
#define BKP 40   // padded LDS row length (halves); 80B rows, 16B aligned

template<bool TRANSB, bool OUTBF16>
__global__ __launch_bounds__(256) void gemm_wmma_bf16(
    const unsigned short* __restrict__ A,
    const unsigned short* __restrict__ B,
    void* __restrict__ C,
    int M, int N, int K, float scale)
{
    __shared__ __attribute__((aligned(16))) unsigned short lA[2][BM * BKP];
    __shared__ __attribute__((aligned(16))) unsigned short lB[2][BN * BKP];

    const int tid  = threadIdx.x;
    const int lane = tid & 31;
    const int wave = tid >> 5;
    const int wm   = wave & 1;         // 0..1 : 64-row strip
    const int wn   = wave >> 1;        // 0..3 : 32-col strip
    const int half = lane >> 4;        // 0/1  : which 16-lane half
    const int l16  = lane & 15;

    const int m0 = blockIdx.y * BM;
    const int n0 = blockIdx.x * BN;

    f32x8 acc[8];
#pragma unroll
    for (int i = 0; i < 8; ++i) acc[i] = (f32x8){0.f,0.f,0.f,0.f,0.f,0.f,0.f,0.f};

    // ---- tile staging ------------------------------------------------------
    auto stageA = [&](int buf, int kb) {          // 128 rows x 32 halves: async
#pragma unroll
        for (int it = 0; it < 2; ++it) {
            int c = tid + it * 256, row = c >> 2, col = (c & 3) * 8;
            g2l_async_b128((unsigned)(size_t)&lA[buf][row * BKP + col],
                           A + (size_t)(m0 + row) * K + kb + col);
        }
    };
    auto stageB = [&](int buf, int kb) {
        if (TRANSB) {                             // 128 rows x 32 halves: async
#pragma unroll
            for (int it = 0; it < 2; ++it) {
                int c = tid + it * 256, row = c >> 2, col = (c & 3) * 8;
                g2l_async_b128((unsigned)(size_t)&lB[buf][row * BKP + col],
                               B + (size_t)(n0 + row) * K + kb + col);
            }
        } else {                                  // [K,N]: transpose into [n][k]
#pragma unroll
            for (int it = 0; it < 2; ++it) {
                int c = tid + it * 256, kk = c >> 4, nn = (c & 15) * 8;
                u32x4 d = *reinterpret_cast<const u32x4*>(
                    B + (size_t)(kb + kk) * N + n0 + nn);
                const unsigned short* ds = reinterpret_cast<const unsigned short*>(&d);
#pragma unroll
                for (int j = 0; j < 8; ++j) lB[buf][(nn + j) * BKP + kk] = ds[j];
            }
        }
    };

    stageA(0, 0); stageB(0, 0);
    const int nsteps = K / BK;
    for (int s = 0; s < nsteps; ++s) {
        wait_async0();          // this wave's async copies have landed in LDS
        __syncthreads();        // all waves done copying + done reading buf we overwrite
        if (s + 1 < nsteps) { stageA((s + 1) & 1, (s + 1) * BK);
                              stageB((s + 1) & 1, (s + 1) * BK); }
        const int buf = s & 1;

        // ---- fragments per ISA VGPR layouts
        ABFrag a[4], b[2];
#pragma unroll
        for (int t = 0; t < 4; ++t) {
            int ar = wm * 64 + t * 16 + l16;   // A: lane -> row, K-chunks at half*8, +16
            a[t].u[0] = *reinterpret_cast<const u32x4*>(&lA[buf][ar * BKP + half * 8]);
            a[t].u[1] = *reinterpret_cast<const u32x4*>(&lA[buf][ar * BKP + half * 8 + 16]);
        }
#pragma unroll
        for (int t = 0; t < 2; ++t) {
            int br = wn * 32 + t * 16 + l16;   // B: lane -> col, K-chunks at half*8, +16
            b[t].u[0] = *reinterpret_cast<const u32x4*>(&lB[buf][br * BKP + half * 8]);
            b[t].u[1] = *reinterpret_cast<const u32x4*>(&lB[buf][br * BKP + half * 8 + 16]);
        }
#pragma unroll
        for (int mt = 0; mt < 4; ++mt)
#pragma unroll
            for (int nt = 0; nt < 2; ++nt)
                acc[mt * 2 + nt] = __builtin_amdgcn_wmma_f32_16x16x32_bf16(
                    false, a[mt].bf, false, b[nt].bf,
                    (short)0, acc[mt * 2 + nt], false, false);
    }

    // ---- store: C layout VGPR r -> (m = r + 8*half, n = l16)
#pragma unroll
    for (int mt = 0; mt < 4; ++mt) {
#pragma unroll
        for (int nt = 0; nt < 2; ++nt) {
            f32x8 v  = acc[mt * 2 + nt];
            int mrow = m0 + wm * 64 + mt * 16 + half * 8;
            int ncol = n0 + wn * 32 + nt * 16 + l16;
#pragma unroll
            for (int r = 0; r < 8; ++r) {
                float val  = v[r] * scale;
                size_t idx = (size_t)(mrow + r) * N + ncol;
                if (OUTBF16) ((unsigned short*)C)[idx] = f2bf(val);
                else         ((float*)C)[idx]          = val;
            }
        }
    }
}

// ------------------------------ softmax ------------------------------------
// One row per wave (wave32), 8 rows per 256-thread block. f32 in, bf16 out.
__global__ __launch_bounds__(256) void softmax_rows(const float* __restrict__ S,
                                                    unsigned short* __restrict__ P,
                                                    int ncols) {
    int row  = blockIdx.x * 8 + (threadIdx.x >> 5);
    int lane = threadIdx.x & 31;
    const float* r = S + (size_t)row * ncols;

    float mx = -3.402823466e38f;
    for (int c = lane; c < ncols; c += 32) mx = fmaxf(mx, r[c]);
#pragma unroll
    for (int off = 16; off; off >>= 1) mx = fmaxf(mx, __shfl_xor(mx, off, 32));

    float sum = 0.f;
    for (int c = lane; c < ncols; c += 32) sum += __expf(r[c] - mx);
#pragma unroll
    for (int off = 16; off; off >>= 1) sum += __shfl_xor(sum, off, 32);
    float inv = 1.0f / sum;

    unsigned short* pr = P + (size_t)row * ncols;
    for (int c = lane; c < ncols; c += 32) pr[c] = f2bf(__expf(r[c] - mx) * inv);
}

// ------------------------------ launcher -----------------------------------
extern "C" void kernel_launch(void* const* d_in, const int* in_sizes, int n_in,
                              void* d_out, int out_size, void* d_ws, size_t ws_size,
                              hipStream_t stream) {
    constexpr int B = 4, S = 2048, D = 1024;
    constexpr size_t XE = (size_t)B * S * D;   // 8 Mi elements
    constexpr size_t WE = (size_t)D * D;       // 1 Mi elements
    constexpr size_t SE = (size_t)S * S;       // 4 Mi elements (per batch)

    const float* x  = (const float*)d_in[0];
    const float* wq = (const float*)d_in[1];
    const float* wk = (const float*)d_in[2];
    const float* wv = (const float*)d_in[3];
    const float* wo = (const float*)d_in[4];

    char* ws = (char*)d_ws;
    unsigned short* xbf   = (unsigned short*)(ws);                 // 16MB, reused as ctx
    unsigned short* wqb   = (unsigned short*)(ws + (XE * 2));
    unsigned short* wkb   = wqb + WE;
    unsigned short* wvb   = wkb + WE;
    unsigned short* wob   = wvb + WE;
    unsigned short* qb    = wob + WE;                              // 16MB
    unsigned short* kbf   = qb  + XE;                              // 16MB
    unsigned short* vb    = kbf + XE;                              // 16MB
    float*          sc    = (float*)(vb + XE);                     // 16MB (per batch)
    unsigned short* attn  = (unsigned short*)(sc + SE);            // 8MB
    unsigned short* ctxbf = xbf;                                   // alias: x dead after QKV

    // 1) fp32 -> bf16 conversions
    cvt_f32_bf16<<<(int)(XE / 256), 256, 0, stream>>>(x,  xbf, (int)XE);
    cvt_f32_bf16<<<(int)(WE / 256), 256, 0, stream>>>(wq, wqb, (int)WE);
    cvt_f32_bf16<<<(int)(WE / 256), 256, 0, stream>>>(wk, wkb, (int)WE);
    cvt_f32_bf16<<<(int)(WE / 256), 256, 0, stream>>>(wv, wvb, (int)WE);
    cvt_f32_bf16<<<(int)(WE / 256), 256, 0, stream>>>(wo, wob, (int)WE);

    // 2) projections: q/k/v = x @ W^T   (M=8192, N=1024, K=1024)
    dim3 gp(D / BN, (B * S) / BM);
    gemm_wmma_bf16<true, true><<<gp, 256, 0, stream>>>(xbf, wqb, qb,  B * S, D, D, 1.0f);
    gemm_wmma_bf16<true, true><<<gp, 256, 0, stream>>>(xbf, wkb, kbf, B * S, D, D, 1.0f);
    gemm_wmma_bf16<true, true><<<gp, 256, 0, stream>>>(xbf, wvb, vb,  B * S, D, D, 1.0f);

    // 3) per-batch attention
    for (int b = 0; b < B; ++b) {
        const unsigned short* qb_b = qb  + (size_t)b * S * D;
        const unsigned short* kb_b = kbf + (size_t)b * S * D;
        const unsigned short* vb_b = vb  + (size_t)b * S * D;
        unsigned short*       cx_b = ctxbf + (size_t)b * S * D;

        // scores = q k^T / sqrt(D)   (M=N=2048, K=1024) -> f32
        dim3 gs(S / BN, S / BM);
        gemm_wmma_bf16<true, false><<<gs, 256, 0, stream>>>(
            qb_b, kb_b, sc, S, S, D, 0.03125f);

        // softmax rows -> bf16 attn
        softmax_rows<<<S / 8, 256, 0, stream>>>(sc, attn, S);

        // ctx = attn @ v   (M=2048, N=1024, K=2048), B stored [K,N]
        dim3 gc(D / BN, S / BM);
        gemm_wmma_bf16<false, true><<<gc, 256, 0, stream>>>(
            attn, vb_b, cx_b, S, D, S, 1.0f);
    }

    // 4) out = ctx @ wo^T  (M=8192, N=1024, K=1024) -> f32 d_out
    gemm_wmma_bf16<true, false><<<gp, 256, 0, stream>>>(
        ctxbf, wob, (float*)d_out, B * S, D, D, 1.0f);
}
